// MouseGAT_5849745457190
// MI455X (gfx1250) — matrix operand order repaired and checked by reading, not compiled, not run on hardware
//
#include <hip/hip_runtime.h>
#include <hip/hip_bf16.h>
#include <hip/hip_fp16.h>

typedef __attribute__((ext_vector_type(16))) _Float16 v16h;
typedef __attribute__((ext_vector_type(8)))  float    v8f;

#define N_NODES   50000
#define N_EDGES   800000
#define E_TOT     850000      // edges + one self-loop per node
#define IN_DIM    128
#define HID       128         // HEADS*HIDDEN
#define HEADS     2
#define HIDDEN    64
#define N_GRAPHS  8
#define N_LATENT  64
#define NEG_SLOPE 0.2f

// ---------------- W -> f16, pre-permuted into per-lane B-fragment layout
// Wfrag half index: ((ntile*4 + kc)*32 + lane)*16 + e
//   lane16 = lane&15, hi = lane>>4
//   source = W[(kc*32 + e + hi*16)*128 + ntile*16 + lane16]
__global__ __launch_bounds__(256) void conv_w_frag(const float* __restrict__ W,
                                                   _Float16* __restrict__ Wfrag) {
    int t = blockIdx.x * 256 + threadIdx.x;   // 16384 halves
    if (t >= IN_DIM * HID) return;
    int ntile = t >> 11;            // 8 n-tiles, 2048 halves each
    int rem   = t & 2047;
    int kc    = rem >> 9;           // 4 K-chunks, 512 halves each
    int rem2  = rem & 511;
    int lane  = rem2 >> 4;          // 32 lanes
    int e     = rem2 & 15;          // 16 halves per lane
    int lane16 = lane & 15;
    int hi     = lane >> 4;
    int k = kc * 32 + e + hi * 16;
    int n = ntile * 16 + lane16;
    Wfrag[t] = (_Float16)W[k * 128 + n];
}

// ---------------------------------------------------- h = x @ W  via WMMA
// block = 256 threads = 8 waves; block covers rows [m0, m0+16), all 128 cols.
__global__ __launch_bounds__(256) void gemm_wmma(const float* __restrict__ x,
                                                 const _Float16* __restrict__ Wfrag,
                                                 float* __restrict__ h) {
    __shared__ _Float16 lds[16 * 128];        // 4 KB x-tile in f16
    const int m0 = blockIdx.x * 16;
    const int t  = threadIdx.x;

    // cooperative load + convert: 2048 f32 -> f16, 8 per thread (contiguous)
    const float* xt = x + (size_t)m0 * 128;
#pragma unroll
    for (int j = 0; j < 8; ++j) {
        int idx = t * 8 + j;
        lds[idx] = (_Float16)xt[idx];
    }
    __syncthreads();

    const int wave   = t >> 5;
    const int lane   = t & 31;
    const int lane16 = lane & 15;
    const int hi     = lane >> 4;             // 0 for lanes 0-15, 1 for 16-31

    const v16h* __restrict__ Wf = (const v16h*)Wfrag;  // 32B-aligned fragments

    v8f acc = {};
#pragma unroll
    for (int kc = 0; kc < 4; ++kc) {          // K = 128 in chunks of 32
        v16h a;
#pragma unroll
        for (int e = 0; e < 16; ++e) {
            // A fragment: row M = lane16, K = hi*16 + e contiguous, matching Wfrag (vectorizes to ds_load_b128)
            int ka = kc * 32 + e + hi * 16;
            a[e] = lds[lane16 * 128 + ka];
        }
        // B fragment: one contiguous 32-byte vector load per lane
        v16h b = Wf[(wave * 4 + kc) * 32 + lane];
        acc = __builtin_amdgcn_wmma_f32_16x16x32_f16(
            /*neg_a=*/false, a, /*neg_b=*/false, b,
            /*c_mod=*/(short)0, acc, /*reuse_a=*/false, /*reuse_b=*/false);
    }

#pragma unroll
    for (int r = 0; r < 8; ++r) {
        int M = m0 + r + hi * 8;
        h[(size_t)M * 128 + wave * 16 + lane16] = acc[r];
    }
}

// ------------------------------------------- per-node attention coefficients
__global__ __launch_bounds__(256) void attn_coef(const float* __restrict__ h,
                                                 const float* __restrict__ att_src,
                                                 const float* __restrict__ att_dst,
                                                 float* __restrict__ a_src,
                                                 float* __restrict__ a_dst) {
    int n = blockIdx.x * 256 + threadIdx.x;
    if (n >= N_NODES) return;
    const float* hr = h + (size_t)n * 128;
#pragma unroll
    for (int hd = 0; hd < HEADS; ++hd) {
        float s = 0.f, d = 0.f;
#pragma unroll 8
        for (int c = 0; c < HIDDEN; ++c) {
            float v = hr[hd * 64 + c];
            s += v * att_src[hd * 64 + c];
            d += v * att_dst[hd * 64 + c];
        }
        a_src[n * 2 + hd] = s;
        a_dst[n * 2 + hd] = d;
    }
}

// --------------------------------------------------------------- zero fill
__global__ __launch_bounds__(256) void zero_f32(float* __restrict__ p, int n) {
    for (int i = blockIdx.x * 256 + threadIdx.x; i < n; i += gridDim.x * 256)
        p[i] = 0.f;
}

// ------------------------- edge pass 1: exp(leaky(logit)) + denom scatter
__global__ __launch_bounds__(256) void edge_pass1(const int* __restrict__ ei,
                                                  const float* __restrict__ a_src,
                                                  const float* __restrict__ a_dst,
                                                  float* __restrict__ wbuf,
                                                  float* __restrict__ denom) {
    int e = blockIdx.x * 256 + threadIdx.x;
    if (e >= E_TOT) return;
    int s, d;
    if (e < N_EDGES) { s = ei[e]; d = ei[N_EDGES + e]; }
    else             { s = d = e - N_EDGES; }          // self-loop
#pragma unroll
    for (int hd = 0; hd < HEADS; ++hd) {
        float l = a_src[s * 2 + hd] + a_dst[d * 2 + hd];
        l = (l >= 0.f) ? l : NEG_SLOPE * l;
        float w = __expf(l);                 // softmax shift-invariant; |l| small
        wbuf[e * 2 + hd] = w;
        atomicAdd(&denom[d * 2 + hd], w);
    }
}

// ------------------- edge pass 2: out[dst] += alpha * h[src], wave per edge
__global__ __launch_bounds__(256) void edge_pass2(const int* __restrict__ ei,
                                                  const float* __restrict__ h,
                                                  const float* __restrict__ wbuf,
                                                  const float* __restrict__ denom,
                                                  float* __restrict__ out_acc) {
    long long tid = (long long)blockIdx.x * 256 + threadIdx.x;
    if (tid >= (long long)E_TOT * 32) return;
    int e    = (int)(tid >> 5);
    int lane = (int)(tid & 31);
    int s, d;
    if (e < N_EDGES) { s = ei[e]; d = ei[N_EDGES + e]; }
    else             { s = d = e - N_EDGES; }
    int c0 = lane * 4;
    int hd = c0 >> 6;
    float alpha = wbuf[e * 2 + hd] / denom[d * 2 + hd];
    float4 hv = *(const float4*)(h + (size_t)s * 128 + c0);
    float* op = out_acc + (size_t)d * 128 + c0;
    atomicAdd(op + 0, alpha * hv.x);
    atomicAdd(op + 1, alpha * hv.y);
    atomicAdd(op + 2, alpha * hv.z);
    atomicAdd(op + 3, alpha * hv.w);
}

// --------------------------- tanh(+bias) then global mean-pool accumulation
__global__ __launch_bounds__(256) void node_pool(const float* __restrict__ out_acc,
                                                 const float* __restrict__ b_gat,
                                                 const int* __restrict__ batch,
                                                 float* __restrict__ pooled,
                                                 float* __restrict__ counts) {
    long long tid = (long long)blockIdx.x * 256 + threadIdx.x;
    if (tid >= (long long)N_NODES * 32) return;
    int n    = (int)(tid >> 5);
    int lane = (int)(tid & 31);
    int g  = batch[n];
    int c0 = lane * 4;
    float4 v = *(const float4*)(out_acc + (size_t)n * 128 + c0);
    atomicAdd(&pooled[g * 128 + c0 + 0], tanhf(v.x + b_gat[c0 + 0]));
    atomicAdd(&pooled[g * 128 + c0 + 1], tanhf(v.y + b_gat[c0 + 1]));
    atomicAdd(&pooled[g * 128 + c0 + 2], tanhf(v.z + b_gat[c0 + 2]));
    atomicAdd(&pooled[g * 128 + c0 + 3], tanhf(v.w + b_gat[c0 + 3]));
    if (lane == 0) atomicAdd(&counts[g], 1.0f);
}

// -------------------------------------- final tiny FC: [8,128]x[64,128]^T
__global__ __launch_bounds__(512) void final_fc(const float* __restrict__ pooled,
                                                const float* __restrict__ counts,
                                                const float* __restrict__ W_fc,
                                                const float* __restrict__ b_fc,
                                                float* __restrict__ out) {
    int t = threadIdx.x;                   // 512 = 8 graphs * 64 latents
    int g = t >> 6, l = t & 63;
    float cnt = counts[g];
    float inv = 1.f / (cnt > 1.f ? cnt : 1.f);
    float s = 0.f;
#pragma unroll 8
    for (int c = 0; c < 128; ++c)
        s += pooled[g * 128 + c] * inv * W_fc[l * 128 + c];
    out[g * 64 + l] = tanhf(s + b_fc[l]);
}

// ---------------------------------------------------------------- launcher
extern "C" void kernel_launch(void* const* d_in, const int* in_sizes, int n_in,
                              void* d_out, int out_size, void* d_ws, size_t ws_size,
                              hipStream_t stream) {
    const float* x        = (const float*)d_in[0];
    const int*   ei       = (const int*)  d_in[1];   // [2, N_EDGES]
    const int*   batch    = (const int*)  d_in[2];
    const float* W_gat    = (const float*)d_in[3];   // [128,128]
    const float* att_src  = (const float*)d_in[4];   // [2,64]
    const float* att_dst  = (const float*)d_in[5];
    const float* b_gat    = (const float*)d_in[6];   // [128]
    const float* W_fc     = (const float*)d_in[7];   // [64,128]
    const float* b_fc     = (const float*)d_in[8];
    float*       out      = (float*)d_out;           // [8,64]

    // workspace layout (floats); zero-region is contiguous at the front
    float* ws       = (float*)d_ws;
    float* out_acc  = ws;                              // N*128   = 6,400,000
    float* denom    = out_acc + (size_t)N_NODES * 128; // N*2     =   100,000
    float* pooled   = denom   + (size_t)N_NODES * 2;   // 8*128   =     1,024
    float* counts   = pooled  + N_GRAPHS * 128;        //               8
    const int ZERO_N = N_NODES * 128 + N_NODES * 2 + N_GRAPHS * 128 + N_GRAPHS;
    float* h        = counts  + N_GRAPHS;              // N*128
    float* a_src    = h       + (size_t)N_NODES * 128; // N*2
    float* a_dst    = a_src   + (size_t)N_NODES * 2;   // N*2
    float* wbuf     = a_dst   + (size_t)N_NODES * 2;   // E_TOT*2
    _Float16* Wfrag = (_Float16*)(wbuf + (size_t)E_TOT * 2);  // 16384 halves, 32B-aligned

    conv_w_frag<<<(IN_DIM * HID + 255) / 256, 256, 0, stream>>>(W_gat, Wfrag);
    zero_f32<<<4096, 256, 0, stream>>>(ws, ZERO_N);

    gemm_wmma<<<N_NODES / 16, 256, 0, stream>>>(x, Wfrag, h);   // 3125 blocks

    attn_coef<<<(N_NODES + 255) / 256, 256, 0, stream>>>(h, att_src, att_dst,
                                                         a_src, a_dst);

    edge_pass1<<<(E_TOT + 255) / 256, 256, 0, stream>>>(ei, a_src, a_dst,
                                                        wbuf, denom);

    long long e2 = (long long)E_TOT * 32;
    edge_pass2<<<(unsigned)((e2 + 255) / 256), 256, 0, stream>>>(ei, h, wbuf,
                                                                 denom, out_acc);

    long long np = (long long)N_NODES * 32;
    node_pool<<<(unsigned)((np + 255) / 256), 256, 0, stream>>>(out_acc, b_gat,
                                                                batch, pooled, counts);

    final_fc<<<1, 512, 0, stream>>>(pooled, counts, W_fc, b_fc, out);
}
